// KMeansLoss_80470507258387
// MI455X (gfx1250) — matrix operand order repaired and checked by reading, not compiled, not run on hardware
//
#include <hip/hip_runtime.h>

#define NUM_CLUSTERS 512
#define EMBED_DIM    256
#define BATCH        65536
#define ALPHA        0.05f

#define WAVES_PER_BLOCK 4
#define NTHREADS        (WAVES_PER_BLOCK * 32)                 // 128
#define ROWS_PER_WAVE   32
#define ROWS_PER_BLOCK  (WAVES_PER_BLOCK * ROWS_PER_WAVE)      // 128
#define KTILE           16
#define NKT             (NUM_CLUSTERS / KTILE)                 // 32
#define NBLOCKS         (BATCH / ROWS_PER_BLOCK)               // 512
// LDS row stride padded by 8 bf16 (16B): 528B row stride == 4 dword-banks (mod 64)
#define ESTRIDE         (EMBED_DIM + 8)                        // 264

#define EPLANE_BYTES    (ROWS_PER_BLOCK * ESTRIDE * 2)         // 67584
#define CPLANE_BYTES    (KTILE * ESTRIDE * 2)                  // 8448
#define SRAW_BYTES      (2 * EPLANE_BYTES)                     // 135168 >= 4*CPLANE_BYTES

typedef __attribute__((ext_vector_type(16))) __bf16 v16bf;
typedef __attribute__((ext_vector_type(8)))  __bf16 v8bf;
typedef __attribute__((ext_vector_type(4)))  __bf16 v4bf;
typedef __attribute__((ext_vector_type(8)))  float  v8f;

struct bf2 { __bf16 h, l; };
__device__ __forceinline__ bf2 bf16_split(float x) {
    __bf16 h = (__bf16)x;
    float  r = x - (float)h;
    bf2 o; o.h = h; o.l = (__bf16)r; return o;
}

#define SPLIT4(v, h, l)                                   \
    do {                                                  \
        bf2 s0 = bf16_split((v).x); h[0] = s0.h; l[0] = s0.l; \
        bf2 s1 = bf16_split((v).y); h[1] = s1.h; l[1] = s1.l; \
        bf2 s2 = bf16_split((v).z); h[2] = s2.h; l[2] = s2.l; \
        bf2 s3 = bf16_split((v).w); h[3] = s3.h; l[3] = s3.l; \
    } while (0)

#define SHUF16(a, b) __builtin_shufflevector(a, b, 0,1,2,3,4,5,6,7,8,9,10,11,12,13,14,15)
#define WMMA_BF16(A, B, C) \
    __builtin_amdgcn_wmma_f32_16x16x32_bf16(false, (A), false, (B), (short)0, (C), false, false)

__global__ __launch_bounds__(NTHREADS)
void kmeans_loss_wmma(const float* __restrict__ emb,
                      const float* __restrict__ cen,
                      float* __restrict__ partial,
                      int atomicMode)
{
    __shared__ __align__(16) unsigned char sRaw[SRAW_BYTES];
    __shared__ float sCsq[NUM_CLUSTERS];
    __shared__ float sMin[WAVES_PER_BLOCK][32];
    __shared__ float sPart[WAVES_PER_BLOCK];

    __bf16* sEhi = reinterpret_cast<__bf16*>(sRaw);
    __bf16* sElo = reinterpret_cast<__bf16*>(sRaw + EPLANE_BYTES);

    const int tid  = threadIdx.x;
    const int wave = tid >> 5;
    const int lane = tid & 31;
    const int l15  = lane & 15;
    const int half = lane >> 4;
    const int rowStart = blockIdx.x * ROWS_PER_BLOCK;

    // ---- stage embedding tile once: fp32 -> bf16 hi + lo planes in LDS ----
    {
        const float4* ev = reinterpret_cast<const float4*>(emb + (size_t)rowStart * EMBED_DIM);
        constexpr int iters = (ROWS_PER_BLOCK * EMBED_DIM / 4) / NTHREADS;   // 64
        for (int i = 0; i < iters; ++i) {
            int f4  = i * NTHREADS + tid;
            int row = f4 >> 6;            // 64 float4 per row
            int c4  = f4 & 63;
            float4 v = ev[f4];
            v4bf h, l;
            SPLIT4(v, h, l);
            *reinterpret_cast<v4bf*>(&sEhi[row * ESTRIDE + c4 * 4]) = h;
            *reinterpret_cast<v4bf*>(&sElo[row * ESTRIDE + c4 * 4]) = l;
        }
    }

    // ---- per-cluster squared norms (fp32, once per block) ----
    for (int c = 0; c < NUM_CLUSTERS / NTHREADS; ++c) {
        int cl = c * NTHREADS + tid;
        const float4* cp = reinterpret_cast<const float4*>(cen + (size_t)cl * EMBED_DIM);
        float s = 0.f;
        for (int j = 0; j < EMBED_DIM / 4; ++j) {
            float4 v = cp[j];
            s += v.x * v.x + v.y * v.y + v.z * v.z + v.w * v.w;
        }
        sCsq[cl] = s;
    }

    // ---- per-row squared norms (fp32): two 16-row subtiles per wave ----
    float xsqA, xsqB;
    {
        int growA = rowStart + wave * ROWS_PER_WAVE + l15;
        int growB = growA + 16;
        const float4* xpA = reinterpret_cast<const float4*>(
            emb + (size_t)growA * EMBED_DIM + half * (EMBED_DIM / 2));
        const float4* xpB = reinterpret_cast<const float4*>(
            emb + (size_t)growB * EMBED_DIM + half * (EMBED_DIM / 2));
        float sA = 0.f, sB = 0.f;
        for (int j = 0; j < EMBED_DIM / 8; ++j) {
            float4 a = xpA[j], b = xpB[j];
            sA += a.x*a.x + a.y*a.y + a.z*a.z + a.w*a.w;
            sB += b.x*b.x + b.y*b.y + b.z*b.z + b.w*b.w;
        }
        sA += __shfl_xor(sA, 16, 32);
        sB += __shfl_xor(sB, 16, 32);
        xsqA = sA; xsqB = sB;
    }

    __syncthreads();

    // ---- preload this wave's A fragments (32 rows, hi+lo) into registers ----
    v16bf Ahi[2][8], Alo[2][8];
    #pragma unroll
    for (int s = 0; s < 2; ++s) {
        const __bf16* eh = &sEhi[(wave * ROWS_PER_WAVE + s * 16 + l15) * ESTRIDE];
        const __bf16* el = &sElo[(wave * ROWS_PER_WAVE + s * 16 + l15) * ESTRIDE];
        #pragma unroll
        for (int dc = 0; dc < 8; ++dc) {
            const int d0 = dc * 32;
            v8bf x0 = *reinterpret_cast<const v8bf*>(eh + d0 + 8 * half);
            v8bf x1 = *reinterpret_cast<const v8bf*>(eh + d0 + 8 * half + 16);
            Ahi[s][dc] = SHUF16(x0, x1);
            v8bf y0 = *reinterpret_cast<const v8bf*>(el + d0 + 8 * half);
            v8bf y1 = *reinterpret_cast<const v8bf*>(el + d0 + 8 * half + 16);
            Alo[s][dc] = SHUF16(y0, y1);
        }
    }
    __syncthreads();   // all E reads done; region can be reused for center tiles

    // ---- double-buffered center tiles alias the (now dead) E region ----
    // buffer b occupies [b*2*CPLANE, b*2*CPLANE + 2*CPLANE): hi plane then lo plane.
    // Selected by integer byte offset (no pointer tables over LDS!).

    // stage tile 0 into buffer 0
    {
        const float4* cv = reinterpret_cast<const float4*>(cen);
        __bf16* dhi = reinterpret_cast<__bf16*>(sRaw);
        __bf16* dlo = reinterpret_cast<__bf16*>(sRaw + CPLANE_BYTES);
        #pragma unroll
        for (int i = 0; i < (KTILE * EMBED_DIM / 4) / NTHREADS; ++i) {   // 8
            int f4  = i * NTHREADS + tid;
            int row = f4 >> 6;
            int c4  = f4 & 63;
            float4 v = cv[f4];
            v4bf h, l;
            SPLIT4(v, h, l);
            *reinterpret_cast<v4bf*>(&dhi[row * ESTRIDE + c4 * 4]) = h;
            *reinterpret_cast<v4bf*>(&dlo[row * ESTRIDE + c4 * 4]) = l;
        }
    }

    float minq0[8], minq1[8];
    #pragma unroll
    for (int i = 0; i < 8; ++i) { minq0[i] = 3.4e38f; minq1[i] = 3.4e38f; }

    __syncthreads();

    for (int kt = 0; kt < NKT; ++kt) {
        const unsigned curOff = (unsigned)(kt & 1) * (2u * CPLANE_BYTES);
        const unsigned nxtOff = (2u * CPLANE_BYTES) - curOff;

        // prefetch next tile's fp32 into registers (hides under WMMA block)
        float4 pre[8];
        if (kt + 1 < NKT) {
            const float4* cv = reinterpret_cast<const float4*>(
                cen + (size_t)(kt + 1) * KTILE * EMBED_DIM);
            #pragma unroll
            for (int i = 0; i < 8; ++i) pre[i] = cv[i * NTHREADS + tid];
        }

        const float csq = sCsq[kt * KTILE + l15];
        v8f acc0 = {};
        v8f acc1 = {};
        const __bf16* bh = reinterpret_cast<const __bf16*>(sRaw + curOff) + l15 * ESTRIDE;
        const __bf16* bl = reinterpret_cast<const __bf16*>(sRaw + curOff + CPLANE_BYTES) + l15 * ESTRIDE;

        #pragma unroll
        for (int dc = 0; dc < 8; ++dc) {
            const int d0 = dc * 32;
            v8bf b0 = *reinterpret_cast<const v8bf*>(bh + d0 + 8 * half);
            v8bf b1 = *reinterpret_cast<const v8bf*>(bh + d0 + 8 * half + 16);
            v16bf Bhi = SHUF16(b0, b1);
            v8bf b2 = *reinterpret_cast<const v8bf*>(bl + d0 + 8 * half);
            v8bf b3 = *reinterpret_cast<const v8bf*>(bl + d0 + 8 * half + 16);
            v16bf Blo = SHUF16(b2, b3);

            // cross ~= hi*hi + hi*lo + lo*hi, two 16-row subtiles share each B
            acc0 = WMMA_BF16(Ahi[0][dc], Bhi, acc0);
            acc1 = WMMA_BF16(Ahi[1][dc], Bhi, acc1);
            acc0 = WMMA_BF16(Ahi[0][dc], Blo, acc0);
            acc1 = WMMA_BF16(Ahi[1][dc], Blo, acc1);
            acc0 = WMMA_BF16(Alo[0][dc], Bhi, acc0);
            acc1 = WMMA_BF16(Alo[1][dc], Bhi, acc1);
        }

        // split + store the prefetched tile into the other buffer
        if (kt + 1 < NKT) {
            __bf16* dhi = reinterpret_cast<__bf16*>(sRaw + nxtOff);
            __bf16* dlo = reinterpret_cast<__bf16*>(sRaw + nxtOff + CPLANE_BYTES);
            #pragma unroll
            for (int i = 0; i < 8; ++i) {
                int f4  = i * NTHREADS + tid;
                int row = f4 >> 6;
                int c4  = f4 & 63;
                v4bf h, l;
                SPLIT4(pre[i], h, l);
                *reinterpret_cast<v4bf*>(&dhi[row * ESTRIDE + c4 * 4]) = h;
                *reinterpret_cast<v4bf*>(&dlo[row * ESTRIDE + c4 * 4]) = l;
            }
        }

        #pragma unroll
        for (int i = 0; i < 8; ++i) {
            minq0[i] = fminf(minq0[i], csq - 2.f * acc0[i]);
            minq1[i] = fminf(minq1[i], csq - 2.f * acc1[i]);
        }

        __syncthreads();   // next buffer fully written / current fully read
    }

    // ---- reduce min across the 16 columns held by each 16-lane half ----
    #pragma unroll
    for (int m = 1; m <= 8; m <<= 1) {
        #pragma unroll
        for (int i = 0; i < 8; ++i) {
            minq0[i] = fminf(minq0[i], __shfl_xor(minq0[i], m, 16));
            minq1[i] = fminf(minq1[i], __shfl_xor(minq1[i], m, 16));
        }
    }
    // C layout: lanes 0-15 VGPR i -> row i ; lanes 16-31 VGPR i -> row i+8
    if (lane == 0) {
        #pragma unroll
        for (int i = 0; i < 8; ++i) {
            sMin[wave][i]      = minq0[i];
            sMin[wave][16 + i] = minq1[i];
        }
    } else if (lane == 16) {
        #pragma unroll
        for (int i = 0; i < 8; ++i) {
            sMin[wave][8 + i]  = minq0[i];
            sMin[wave][24 + i] = minq1[i];
        }
    }
    __syncthreads();

    float val = 0.f;
    if (lane < 16) {
        float dA = xsqA + sMin[wave][lane];        // xsqA in lane l<16 is row l
        float dB = xsqB + sMin[wave][16 + lane];   // xsqB is row 16+l
        val = sqrtf(fmaxf(dA, 0.f)) + sqrtf(fmaxf(dB, 0.f));
    }
    #pragma unroll
    for (int m = 1; m < 32; m <<= 1) val += __shfl_xor(val, m, 32);
    if (lane == 0) sPart[wave] = val;
    __syncthreads();
    if (tid == 0) {
        float s = 0.f;
        #pragma unroll
        for (int w = 0; w < WAVES_PER_BLOCK; ++w) s += sPart[w];
        if (atomicMode) atomicAdd(partial, s * (ALPHA / (float)BATCH));
        else            partial[blockIdx.x] = s;   // deterministic per-block partial
    }
}

__global__ __launch_bounds__(256)
void kmeans_reduce(const float* __restrict__ partial, float* __restrict__ out, int n)
{
    __shared__ float s[256];
    float v = 0.f;
    for (int i = threadIdx.x; i < n; i += 256) v += partial[i];
    s[threadIdx.x] = v;
    __syncthreads();
    for (int st = 128; st > 0; st >>= 1) {
        if (threadIdx.x < st) s[threadIdx.x] += s[threadIdx.x + st];
        __syncthreads();
    }
    if (threadIdx.x == 0) out[0] = s[0] * (ALPHA / (float)BATCH);
}

__global__ void zero_out(float* out) { if (threadIdx.x == 0) out[0] = 0.f; }

extern "C" void kernel_launch(void* const* d_in, const int* in_sizes, int n_in,
                              void* d_out, int out_size, void* d_ws, size_t ws_size,
                              hipStream_t stream) {
    const float* emb = (const float*)d_in[0];   // [BATCH, EMBED_DIM] fp32
    const float* cen = (const float*)d_in[1];   // [NUM_CLUSTERS, EMBED_DIM] fp32
    float* out = (float*)d_out;

    if (ws_size >= NBLOCKS * sizeof(float)) {
        float* part = (float*)d_ws;
        kmeans_loss_wmma<<<NBLOCKS, NTHREADS, 0, stream>>>(emb, cen, part, 0);
        kmeans_reduce<<<1, 256, 0, stream>>>(part, out, NBLOCKS);
    } else {
        // fallback: atomic accumulation directly into out[0]
        zero_out<<<1, 1, 0, stream>>>(out);
        kmeans_loss_wmma<<<NBLOCKS, NTHREADS, 0, stream>>>(emb, cen, out, 1);
    }
}
